// MolGNN_18803366822157
// MI455X (gfx1250) — matrix operand order repaired and checked by reading, not compiled, not run on hardware
//
#include <hip/hip_runtime.h>
#include <stdint.h>

// ---------------------------------------------------------------------------
// Fused MolGNN forward for MI455X (gfx1250).
//  - 1 workgroup (8 wave32) per batch graph; all activations resident in LDS
//    (~284 KB of the 320 KB WGP pool) for all 3 layers -> adjacency (768 MB
//    total) is the only large HBM stream, read exactly once per layer.
//  - All GEMMs via v_wmma_f32_16x16x32_bf16 (fp32 accumulate).
//  - Adjacency staged cooperatively into a DOUBLE-BUFFERED LDS panel shared
//    by all 8 waves; panel mt+2 prefetched with global_prefetch_b8.
//  - h is stored COLUMN-major (hT[hid][node]) so WMMA D-frags store with one
//    ds_store_b128 and B-frags load with two ds_load_b128 (no scalar LDS ops
//    in the hot adj@h loop).
//  - bf16 conversion via native casts -> hardware v_cvt_pk_bf16_f32.
// ---------------------------------------------------------------------------

constexpr int B_  = 256;   // batch
constexpr int N_  = 256;   // nodes
constexpr int D0_ = 64;    // input feature dim
constexpr int H_  = 128;   // hidden dim
constexpr int O_  = 16;    // output dim
constexpr int L_  = 3;     // layers
constexpr int K_  = 4;     // bond types

constexpr int LDB_ = 136;  // bf16 row stride (elems), x / neigh buffer [node][hid]
constexpr int LDH_ = 264;  // bf16 row stride (elems), hT buffer [hid][node]
constexpr int LDM_ = 132;  // f32 row stride (elems), message buffer [node][hid]
constexpr int LDP_ = 264;  // bf16 row stride (elems), adjacency panels [16][node]

constexpr int OFF_X  = 0;                         // 256*136*2 = 69632
constexpr int OFF_HT = OFF_X  + N_ * LDB_ * 2;    // 69632  (128*264*2 = 67584)
constexpr int OFF_M  = OFF_HT + H_ * LDH_ * 2;    // 137216 (256*132*4 = 135168)
constexpr int OFF_P0 = OFF_M  + N_ * LDM_ * 4;    // 272384 (16*264*2 = 8448)
constexpr int OFF_P1 = OFF_P0 + 16 * LDP_ * 2;    // 280832
constexpr int OFF_EB = OFF_P1 + 16 * LDP_ * 2;    // 289280
constexpr int OFF_G  = OFF_EB + H_ * 4;           // 289792
constexpr int SMEM_BYTES = OFF_G + H_ * 4;        // 290304 (< 320 KB WGP LDS)

typedef __attribute__((ext_vector_type(16))) __bf16 v16bf;
typedef __attribute__((ext_vector_type(8)))  __bf16 bf8v;
typedef __attribute__((ext_vector_type(4)))  __bf16 bf4v;
typedef __attribute__((ext_vector_type(8)))  float  v8f;
typedef __attribute__((ext_vector_type(4)))  float  f4v;

static __device__ __forceinline__ bf4v cvt4(f4v v) {
  return __builtin_convertvector(v, bf4v);      // -> v_cvt_pk_bf16_f32
}

static __device__ __forceinline__ v8f wmma_bf16(v16bf a, v16bf b, v8f c) {
  return __builtin_amdgcn_wmma_f32_16x16x32_bf16(
      /*neg_a=*/false, a, /*neg_b=*/false, b,
      /*c_mod=*/(short)0, c, /*reuse_a=*/false, /*reuse_b=*/false);
}

// A fragment (16x32 bf16, MxK) from a row-major LDS buffer [rows][ld].
// Layout: lanes 0-15 row M=lane, K 0..15; lanes 16-31 K 16..31.
// k0 % 32 == 0 and 16B row alignment hold at all call sites -> two ds_load_b128.
static __device__ __forceinline__ v16bf
load_a_lds(const __bf16* buf, int ld, int row0, int k0, int lane) {
  const int m  = lane & 15;
  const int kb = (lane >> 4) << 4;                        // 0 or 16
  const __bf16* p = buf + (size_t)(row0 + m) * ld + k0 + kb;
  uint4 lo = *(const uint4*)(p);                          // K = kb .. kb+7
  uint4 hi = *(const uint4*)(p + 8);                      // K = kb+8 .. kb+15
  v16bf a;
  __builtin_memcpy(&a, &lo, 16);
  __builtin_memcpy(reinterpret_cast<char*>(&a) + 16, &hi, 16);
  return a;
}

// B fragment (32x16 bf16, KxN) from the COLUMN-major hT buffer [hid][node]:
// element (K=kk, N=n) = hT[n0+n][k0+kk] -> 16 consecutive bf16 per lane.
// Lanes 0-15: N=lane, K 0..15; lanes 16-31: N=lane-16, K 16..31.
static __device__ __forceinline__ v16bf
load_b_hT(const __bf16* bufT, int ldT, int k0, int n0, int lane) {
  const int n  = lane & 15;
  const int kb = (lane >> 4) << 4;                        // 0 or 16
  const __bf16* p = bufT + (size_t)(n0 + n) * ldT + k0 + kb;
  uint4 lo = *(const uint4*)(p);                          // K = kb .. kb+7
  uint4 hi = *(const uint4*)(p + 8);                      // K = kb+8 .. kb+15
  v16bf b;
  __builtin_memcpy(&b, &lo, 16);
  __builtin_memcpy(reinterpret_cast<char*>(&b) + 16, &hi, 16);
  return b;
}

// B fragment for W^T where W is torch-layout [out][in] fp32:
// element (K=kk, N=n) = W[n0+n][k0+kk] -> 16 consecutive floats per lane.
static __device__ __forceinline__ v16bf
load_b_wT(const float* __restrict__ W, int ldw, int k0, int n0, int lane) {
  const int n  = lane & 15;
  const int kb = (lane >> 4) << 4;
  const f4v* p = (const f4v*)(W + (size_t)(n0 + n) * ldw + k0 + kb);
  bf4v q0 = cvt4(p[0]), q1 = cvt4(p[1]), q2 = cvt4(p[2]), q3 = cvt4(p[3]);
  v16bf b;
  __builtin_memcpy(reinterpret_cast<char*>(&b) +  0, &q0, 8);
  __builtin_memcpy(reinterpret_cast<char*>(&b) +  8, &q1, 8);
  __builtin_memcpy(reinterpret_cast<char*>(&b) + 16, &q2, 8);
  __builtin_memcpy(reinterpret_cast<char*>(&b) + 24, &q3, 8);
  return b;
}

__global__ __launch_bounds__(256, 1)
void molgnn_fused_kernel(
    const float* __restrict__ feat,   // [B,N,64]
    const float* __restrict__ adj,    // [B,K,N,N]
    const float* __restrict__ nW0, const float* __restrict__ nb0,
    const float* __restrict__ nW1, const float* __restrict__ nb1,
    const float* __restrict__ nW2, const float* __restrict__ nb2,
    const float* __restrict__ eW,     // [K,H,H]
    const float* __restrict__ ebias,  // [K,H]
    const float* __restrict__ lng,    // [L,H]
    const float* __restrict__ lnbeta, // [L,H]
    const float* __restrict__ oW,     // [O,H]
    const float* __restrict__ ob,     // [O]
    float* __restrict__ out)          // [B,O]
{
  extern __shared__ char smem[];
  __bf16* xbuf  = (__bf16*)(smem + OFF_X);   // x (phase A/C) / neigh (phase B), [node][hid]
  __bf16* hT    = (__bf16*)(smem + OFF_HT);  // h after node linear, [hid][node]
  float*  msg   = (float*) (smem + OFF_M);   // message accumulator f32, [node][hid]
  __bf16* pan0  = (__bf16*)(smem + OFF_P0);  // adjacency panel, buffer 0
  __bf16* pan1  = (__bf16*)(smem + OFF_P1);  // adjacency panel, buffer 1
  float*  ebsum = (float*) (smem + OFF_EB);  // sum_k edge_b[k], f32[128]
  float*  graph = (float*) (smem + OFF_G);   // pooled graph repr, f32[128]

  const int b     = blockIdx.x;
  const int tid   = threadIdx.x;
  const int lane  = tid & 31;
  const int wave  = tid >> 5;            // 8 waves; wave owns hidden cols [n0, n0+16)
  const int n0    = wave * 16;
  const int mlane = lane & 15;           // C/D frag: column within tile
  const int mhalf = (lane >> 4) << 3;    // C/D frag: row base 0 or 8

  // ---- stage input features (fp32 -> bf16, packed 8B stores) + edge-bias sum
  {
    const f4v* f4p = (const f4v*)(feat + (size_t)b * N_ * D0_);
#pragma unroll
    for (int i = 0; i < 16; ++i) {
      int idx = i * 256 + tid;           // 4096 float4 across 256 threads
      f4v v = f4p[idx];
      int flat = idx * 4;
      int row = flat >> 6, col = flat & 63;
      *(bf4v*)(xbuf + (size_t)row * LDB_ + col) = cvt4(v);
    }
    if (tid < H_) {
      float s = 0.f;
#pragma unroll
      for (int k = 0; k < K_; ++k) s += ebias[k * H_ + tid];
      ebsum[tid] = s;
    }
  }
  __syncthreads();

  for (int layer = 0; layer < L_; ++layer) {
    const bool   wide = (layer != 0);
    const int    D    = wide ? H_ : D0_;
    const float* W    = (layer == 0) ? nW0 : (layer == 1) ? nW1 : nW2;
    const float* Wb   = (layer == 0) ? nb0 : (layer == 1) ? nb1 : nb2;

    // ============ Phase A: hT = (x @ W^T + b)^T ============
    {
      v16bf wb0 = load_b_wT(W, D, 0,  n0, lane);
      v16bf wb1 = load_b_wT(W, D, 32, n0, lane);
      v16bf wb2 = {}, wb3 = {};
      if (wide) { wb2 = load_b_wT(W, D, 64, n0, lane);
                  wb3 = load_b_wT(W, D, 96, n0, lane); }
      const float bias = Wb[n0 + mlane];
      for (int mt = 0; mt < 16; ++mt) {
        v8f acc = {};
        acc = wmma_bf16(load_a_lds(xbuf, LDB_, mt * 16, 0,  lane), wb0, acc);
        acc = wmma_bf16(load_a_lds(xbuf, LDB_, mt * 16, 32, lane), wb1, acc);
        if (wide) {
          acc = wmma_bf16(load_a_lds(xbuf, LDB_, mt * 16, 64, lane), wb2, acc);
          acc = wmma_bf16(load_a_lds(xbuf, LDB_, mt * 16, 96, lane), wb3, acc);
        }
        // D-frag: lane = hid col (n0+mlane), vgprs = 8 consecutive node rows
        // -> column-major store is ONE ds_store_b128.
        bf8v hv;
#pragma unroll
        for (int r = 0; r < 8; ++r) hv[r] = (__bf16)(acc[r] + bias);
        *(bf8v*)(hT + (size_t)(n0 + mlane) * LDH_ + mt * 16 + mhalf) = hv;
      }
    }
    __syncthreads();

    // ============ Phase B: messages = sum_k (adj_k @ h) @ edge_W_k^T ========
    v8f macc[16];
#pragma unroll
    for (int mt = 0; mt < 16; ++mt) { v8f z = {}; macc[mt] = z; }

    for (int k = 0; k < K_; ++k) {
      const float* Ak = adj + (((size_t)b * K_ + k) * N_) * N_;
      const float* Ek = eW + (size_t)k * H_ * H_;
      v16bf ew0 = load_b_wT(Ek, H_, 0,  n0, lane);
      v16bf ew1 = load_b_wT(Ek, H_, 32, n0, lane);
      v16bf ew2 = load_b_wT(Ek, H_, 64, n0, lane);
      v16bf ew3 = load_b_wT(Ek, H_, 96, n0, lane);

      // cooperative stage of one 16x256 adjacency row-panel (fp32 -> bf16)
      auto stage_panel = [&](int mt, __bf16* dst) {
        const f4v* a4 = (const f4v*)(Ak + (size_t)mt * 16 * N_);
#pragma unroll
        for (int i = 0; i < 4; ++i) {
          int idx = i * 256 + tid;       // 1024 float4 across 256 threads
          f4v v = a4[idx];
          int flat = idx * 4;
          int row = flat >> 8, col = flat & 255;
          *(bf4v*)(dst + (size_t)row * LDP_ + col) = cvt4(v);
        }
      };

      // --- B1: neigh = adj_k @ h -> xbuf (bf16), double-buffered panels.
      //     Each streamed adjacency byte is consumed by all 8 waves.
      stage_panel(0, pan0);
      __syncthreads();
      for (int mt = 0; mt < 16; ++mt) {
        const __bf16* pcur = (mt & 1) ? pan1 : pan0;
        __bf16*       pnxt = (mt & 1) ? pan0 : pan1;
        if (mt + 2 < 16)  // pull panel mt+2 toward L2 (64B per lane)
          __builtin_prefetch((const char*)(Ak + (size_t)(mt + 2) * 16 * N_) + tid * 64, 0, 1);
        if (mt + 1 < 16) stage_panel(mt + 1, pnxt);   // overlap with compute below
        v8f acc = {};
#pragma unroll
        for (int c = 0; c < 8; ++c) {
          v16bf a  = load_a_lds(pcur, LDP_, 0, c * 32, lane);
          v16bf bh = load_b_hT(hT, LDH_, c * 32, n0, lane);
          acc = wmma_bf16(a, bh, acc);
        }
        __bf16* nr = xbuf + (size_t)(mt * 16 + mhalf) * LDB_ + n0 + mlane;
#pragma unroll
        for (int r = 0; r < 8; ++r) nr[(size_t)r * LDB_] = (__bf16)acc[r];
        __syncthreads();   // next-panel stores + this tile's neigh stores done
      }

      // --- B2: macc += neigh @ edge_W_k^T (per-wave register accumulation)
#pragma unroll
      for (int mt = 0; mt < 16; ++mt) {
        macc[mt] = wmma_bf16(load_a_lds(xbuf, LDB_, mt * 16, 0,  lane), ew0, macc[mt]);
        macc[mt] = wmma_bf16(load_a_lds(xbuf, LDB_, mt * 16, 32, lane), ew1, macc[mt]);
        macc[mt] = wmma_bf16(load_a_lds(xbuf, LDB_, mt * 16, 64, lane), ew2, macc[mt]);
        macc[mt] = wmma_bf16(load_a_lds(xbuf, LDB_, mt * 16, 96, lane), ew3, macc[mt]);
      }
      __syncthreads();
    }

    // spill message accumulators to LDS (f32) for row-wise LayerNorm
#pragma unroll
    for (int mt = 0; mt < 16; ++mt) {
      float* mr = msg + (size_t)(mt * 16 + mhalf) * LDM_ + n0 + mlane;
#pragma unroll
      for (int r = 0; r < 8; ++r) mr[(size_t)r * LDM_] = macc[mt][r];
    }
    __syncthreads();

    // ============ Phase C: x = relu(LN(h + msg + eb_sum)), 1 thread / node ==
    {
      const int row = tid;
      const __bf16* hc0 = hT + row;                    // column of hT
      const f4v*  mr4 = (const f4v*)(msg + (size_t)row * LDM_);
      const f4v*  eb4 = (const f4v*)ebsum;
      float mu = 0.f, s2 = 0.f;
#pragma unroll 4
      for (int q = 0; q < H_ / 4; ++q) {
        f4v hv = { (float)hc0[(size_t)(4 * q + 0) * LDH_],
                   (float)hc0[(size_t)(4 * q + 1) * LDH_],
                   (float)hc0[(size_t)(4 * q + 2) * LDH_],
                   (float)hc0[(size_t)(4 * q + 3) * LDH_] };
        f4v v = hv + mr4[q] + eb4[q];
        mu += v.x + v.y + v.z + v.w;
        s2 += v.x * v.x + v.y * v.y + v.z * v.z + v.w * v.w;
      }
      mu *= (1.f / H_);
      float var  = s2 * (1.f / H_) - mu * mu;
      float rstd = rsqrtf(var + 1e-5f);
      const f4v* g4 = (const f4v*)(lng    + layer * H_);
      const f4v* t4 = (const f4v*)(lnbeta + layer * H_);
      const f4v  zf = {0.f, 0.f, 0.f, 0.f};
#pragma unroll 4
      for (int q = 0; q < H_ / 4; ++q) {
        f4v hv = { (float)hc0[(size_t)(4 * q + 0) * LDH_],
                   (float)hc0[(size_t)(4 * q + 1) * LDH_],
                   (float)hc0[(size_t)(4 * q + 2) * LDH_],
                   (float)hc0[(size_t)(4 * q + 3) * LDH_] };
        f4v v = hv + mr4[q] + eb4[q];
        f4v y = (v - mu) * rstd * g4[q] + t4[q];
        *(bf4v*)(xbuf + (size_t)row * LDB_ + q * 4) =
            cvt4(__builtin_elementwise_max(y, zf));
      }
    }
    __syncthreads();
  }

  // ============ Readout: graph = sum_n x ; out = graph @ out_W^T + out_b ====
  if (tid < H_) {
    float s = 0.f;
    for (int row = 0; row < N_; ++row) s += (float)xbuf[(size_t)row * LDB_ + tid];
    graph[tid] = s;
  }
  __syncthreads();
  if (tid < O_) {
    float acc = ob[tid];
    const float* w = oW + (size_t)tid * H_;
#pragma unroll 16
    for (int hc = 0; hc < H_; ++hc) acc += graph[hc] * w[hc];
    out[(size_t)b * O_ + tid] = acc;
  }
}

extern "C" void kernel_launch(void* const* d_in, const int* in_sizes, int n_in,
                              void* d_out, int out_size, void* d_ws, size_t ws_size,
                              hipStream_t stream) {
  (void)in_sizes; (void)n_in; (void)d_ws; (void)ws_size; (void)out_size;
  const float* feat = (const float*)d_in[0];
  const float* adj  = (const float*)d_in[1];
  const float* nW0  = (const float*)d_in[2];
  const float* nb0  = (const float*)d_in[3];
  const float* nW1  = (const float*)d_in[4];
  const float* nb1  = (const float*)d_in[5];
  const float* nW2  = (const float*)d_in[6];
  const float* nb2  = (const float*)d_in[7];
  const float* eW   = (const float*)d_in[8];
  const float* eb   = (const float*)d_in[9];
  const float* lng  = (const float*)d_in[10];
  const float* lnb  = (const float*)d_in[11];
  const float* oW   = (const float*)d_in[12];
  const float* ob   = (const float*)d_in[13];
  float* out = (float*)d_out;

  static_assert(SMEM_BYTES <= 320 * 1024, "LDS budget exceeded");
  (void)hipFuncSetAttribute(reinterpret_cast<const void*>(molgnn_fused_kernel),
                            hipFuncAttributeMaxDynamicSharedMemorySize, SMEM_BYTES);

  molgnn_fused_kernel<<<dim3(B_), dim3(256), SMEM_BYTES, stream>>>(
      feat, adj, nW0, nb0, nW1, nb1, nW2, nb2, eW, eb, lng, lnb, oW, ob, out);
}
